// DeepSeekMoE_23708219474204
// MI455X (gfx1250) — matrix operand order repaired and checked by reading, not compiled, not run on hardware
//
#include <hip/hip_runtime.h>
#include <hip/hip_bf16.h>
#include <math.h>

#define TOKENS 4096
#define DIMSZ  2048
#define HIDSZ  1024
#define NEXP   16

#define BM 128
#define BN 128
#define BK 32
#define LDSS 40   // LDS row stride in bf16 elements (80B: conflict-breaking, 16B-aligned)

typedef __attribute__((ext_vector_type(16))) __bf16 v16bf;
typedef __attribute__((ext_vector_type(8)))  float  v8f;

union FragAB { v16bf v; uint4 q[2]; };
union Pack8  { uint4 q; __bf16 h[8]; };

__device__ __forceinline__ float fast_silu(float v) {
    // v * sigmoid(v); rcp instead of IEEE divide (bf16 output, precision ample)
    return v * __builtin_amdgcn_rcpf(1.0f + __expf(-v));
}

// ---------------------------------------------------------------------------
// Small kernels
// ---------------------------------------------------------------------------
__global__ void zero_kernel(float* probsum, int* count) {
    int t = threadIdx.x;
    if (t < NEXP) { probsum[t] = 0.0f; count[t] = 0; }
}

__global__ void cvt_x_kernel(const float* __restrict__ x, __bf16* __restrict__ xb) {
    size_t i = ((size_t)blockIdx.x * 256 + threadIdx.x) * 8;
    float4 a = *(const float4*)(x + i);
    float4 b = *(const float4*)(x + i + 4);
    Pack8 p;
    p.h[0] = (__bf16)a.x; p.h[1] = (__bf16)a.y; p.h[2] = (__bf16)a.z; p.h[3] = (__bf16)a.w;
    p.h[4] = (__bf16)b.x; p.h[5] = (__bf16)b.y; p.h[6] = (__bf16)b.z; p.h[7] = (__bf16)b.w;
    *(uint4*)(xb + i) = p.q;
}

// one wave (32 lanes) per token
__global__ void gate_kernel(const float* __restrict__ x, const float* __restrict__ gw,
                            float* probsum, int* count, int* list, float* wlist) {
    int wave = threadIdx.x >> 5;
    int lane = threadIdx.x & 31;
    int t = blockIdx.x * 8 + wave;

    float acc[NEXP];
#pragma unroll
    for (int e = 0; e < NEXP; ++e) acc[e] = 0.0f;

    const float* xr = x + (size_t)t * DIMSZ;
    for (int d = lane; d < DIMSZ; d += 32) {
        float xv = xr[d];
        const float* g = gw + (size_t)d * NEXP;
#pragma unroll
        for (int e = 0; e < NEXP; ++e) acc[e] = fmaf(xv, g[e], acc[e]);
    }
#pragma unroll
    for (int off = 16; off > 0; off >>= 1) {
#pragma unroll
        for (int e = 0; e < NEXP; ++e) acc[e] += __shfl_xor(acc[e], off, 32);
    }
    float mx = acc[0];
#pragma unroll
    for (int e = 1; e < NEXP; ++e) mx = fmaxf(mx, acc[e]);
    float p[NEXP]; float s = 0.0f;
#pragma unroll
    for (int e = 0; e < NEXP; ++e) { p[e] = __expf(acc[e] - mx); s += p[e]; }
    float inv = 1.0f / s;
#pragma unroll
    for (int e = 0; e < NEXP; ++e) p[e] *= inv;
    int i1 = 0;
#pragma unroll
    for (int e = 1; e < NEXP; ++e) if (p[e] > p[i1]) i1 = e;
    int i2 = (i1 == 0) ? 1 : 0;
#pragma unroll
    for (int e = 0; e < NEXP; ++e) if (e != i1 && p[e] > p[i2]) i2 = e;
    float wsum = p[i1] + p[i2];
    float w1n = p[i1] / wsum, w2n = p[i2] / wsum;

    if (lane == 0) {
#pragma unroll
        for (int e = 0; e < NEXP; ++e) atomicAdd(&probsum[e], p[e]);
        int pos = atomicAdd(&count[i1], 1);
        list[i1 * TOKENS + pos] = t; wlist[i1 * TOKENS + pos] = w1n;
        pos = atomicAdd(&count[i2], 1);
        list[i2 * TOKENS + pos] = t; wlist[i2 * TOKENS + pos] = w2n;
    }
}

__global__ void finalize_gate_kernel(const float* probsum, const int* count,
                                     int* offsets, float* aux_out) {
    if (threadIdx.x == 0 && blockIdx.x == 0) {
        int off = 0; float aux = 0.0f;
        for (int e = 0; e < NEXP; ++e) {
            offsets[e] = off; off += count[e];
            float m = probsum[e] / (float)TOKENS;
            aux += m * m;
        }
        *aux_out = aux * (float)NEXP;
    }
}

// ---------------------------------------------------------------------------
// WMMA GEMM core:  C[M x N] = A[M x K](bf16) * B[K x N](fp32, cvt->bf16)
// Block: 256 thr = 8 waves (4x2), tile 128x128, K-step 32. Wave = 32x64 = 2x4 WMMAs.
// EPI: 0 = silu -> bf16 store   1 = f32 store   3 = f32 +=   2 = scaled atomic scatter
// GUARD: row bound check (routed tiles only)
// ---------------------------------------------------------------------------
template <int EPI, bool GUARD>
__device__ __forceinline__ void gemm_core(
    const __bf16* __restrict__ A, const float* __restrict__ B,
    int M, int K, int N,
    const int* __restrict__ gatherA,
    __bf16* __restrict__ outb,
    float* __restrict__ outf,
    const int* __restrict__ scatter,
    const float* __restrict__ scale) {

    __shared__ __bf16 As[BM * LDSS];
    __shared__ __bf16 Bs[BN * LDSS];

    const int tid  = threadIdx.x;
    const int lane = tid & 31;
    const int wave = tid >> 5;
    const int wm = (wave >> 1) * 32;   // 4 wave-rows
    const int wn = (wave & 1) * 64;    // 2 wave-cols
    const int m0 = blockIdx.y * BM;
    const int n0 = blockIdx.x * BN;

    v8f acc[2][4];
#pragma unroll
    for (int i = 0; i < 2; ++i)
#pragma unroll
        for (int j = 0; j < 4; ++j)
#pragma unroll
            for (int r = 0; r < 8; ++r) acc[i][j][r] = 0.0f;

    // A-tile: 128x32 bf16; 2 iters x 256 thr x 8 bf16 (16B)
    const __bf16* aptr[2];
    int adst[2];
#pragma unroll
    for (int it = 0; it < 2; ++it) {
        int idx = tid + it * 256;
        int r = idx >> 2, cg = idx & 3;
        int grow = m0 + r;
        int srow;
        if (gatherA) srow = (grow < M) ? gatherA[grow] : 0;
        else         srow = grow;
        aptr[it] = A + (size_t)srow * K + cg * 8;
        adst[it] = r * LDSS + cg * 8;
    }
    // B-tile: 32(K) x 128(N) fp32; each thread 16 fp32 from one row
    const int bkk = tid >> 3, bncg = tid & 7;
    const float* bptr = B + (size_t)bkk * N + n0 + bncg * 16;

    for (int k0 = 0; k0 < K; k0 += BK) {
        uint4  a0 = *(const uint4*)(aptr[0] + k0);
        uint4  a1 = *(const uint4*)(aptr[1] + k0);
        const float* bp = bptr + (size_t)k0 * N;
        float4 b0 = *(const float4*)(bp);
        float4 b1 = *(const float4*)(bp + 4);
        float4 b2 = *(const float4*)(bp + 8);
        float4 b3 = *(const float4*)(bp + 12);

        __syncthreads();
        *(uint4*)&As[adst[0]] = a0;
        *(uint4*)&As[adst[1]] = a1;
        __bf16 bb[16];
        bb[0]=(__bf16)b0.x;  bb[1]=(__bf16)b0.y;  bb[2]=(__bf16)b0.z;  bb[3]=(__bf16)b0.w;
        bb[4]=(__bf16)b1.x;  bb[5]=(__bf16)b1.y;  bb[6]=(__bf16)b1.z;  bb[7]=(__bf16)b1.w;
        bb[8]=(__bf16)b2.x;  bb[9]=(__bf16)b2.y;  bb[10]=(__bf16)b2.z; bb[11]=(__bf16)b2.w;
        bb[12]=(__bf16)b3.x; bb[13]=(__bf16)b3.y; bb[14]=(__bf16)b3.z; bb[15]=(__bf16)b3.w;
#pragma unroll
        for (int jj = 0; jj < 16; ++jj) Bs[(bncg * 16 + jj) * LDSS + bkk] = bb[jj];
        __syncthreads();

        FragAB af[2], bf[4];
#pragma unroll
        for (int i = 0; i < 2; ++i) {
            int r  = wm + i * 16 + (lane & 15);
            int kh = (lane >> 4) * 16;  // lanes 0-15: K0-15, lanes 16-31: K16-31
            af[i].q[0] = *(const uint4*)&As[r * LDSS + kh];
            af[i].q[1] = *(const uint4*)&As[r * LDSS + kh + 8];
        }
#pragma unroll
        for (int j = 0; j < 4; ++j) {
            int n  = wn + j * 16 + (lane & 15);
            int kb = (lane >> 4) * 16;  // 16 contiguous K from transposed tile
            bf[j].q[0] = *(const uint4*)&Bs[n * LDSS + kb];
            bf[j].q[1] = *(const uint4*)&Bs[n * LDSS + kb + 8];
        }
#pragma unroll
        for (int i = 0; i < 2; ++i)
#pragma unroll
            for (int j = 0; j < 4; ++j)
                acc[i][j] = __builtin_amdgcn_wmma_f32_16x16x32_bf16(
                    false, af[i].v, false, bf[j].v, (short)0, acc[i][j], false, false);
    }

    // Epilogue. C layout: VGPR r -> row (lane/16)*8 + r, col = lane%16.
    // Per-row offsets hoisted out of j-loop (rows shared by all 4 j-subtiles).
#pragma unroll
    for (int i = 0; i < 2; ++i) {
        const int mbase = m0 + wm + i * 16 + (lane >> 4) * 8;
        size_t rowoff[8];
        float  rscale[8];
        bool   rok[8];
#pragma unroll
        for (int r = 0; r < 8; ++r) {
            int m = mbase + r;
            if constexpr (EPI == 2) {
                rok[r] = (m < M);
                int tok = rok[r] ? scatter[m] : 0;
                rscale[r] = rok[r] ? scale[m] : 0.0f;
                rowoff[r] = (size_t)tok * N;
            } else {
                rok[r] = GUARD ? (m < M) : true;
                rowoff[r] = (size_t)m * N;
            }
        }
#pragma unroll
        for (int j = 0; j < 4; ++j) {
            const int n = n0 + wn + j * 16 + (lane & 15);
#pragma unroll
            for (int r = 0; r < 8; ++r) {
                if (GUARD && !rok[r]) continue;
                float v = acc[i][j][r];
                if constexpr (EPI == 0)      outb[rowoff[r] + n] = (__bf16)fast_silu(v);
                else if constexpr (EPI == 1) outf[rowoff[r] + n] = v;
                else if constexpr (EPI == 3) outf[rowoff[r] + n] += v;
                else                         atomicAdd(&outf[rowoff[r] + n], rscale[r] * v);
            }
        }
    }
}

template <int EPI>
__global__ void gemm_dense_kernel(const __bf16* A, const float* B, int M, int K, int N,
                                  __bf16* outb, float* outf) {
    gemm_core<EPI, false>(A, B, M, K, N, nullptr, outb, outf, nullptr, nullptr);
}

template <int EPI>
__global__ void gemm_routed_kernel(const __bf16* Abase, const float* Wbase, int K, int N,
                                   const int* count, const int* offsets,
                                   const int* list, const float* wlist,
                                   __bf16* hr, float* outf) {
    int e = blockIdx.z;
    int M = count[e];
    if ((int)blockIdx.y * BM >= M) return;
    const float* B = Wbase + (size_t)e * K * N;
    const int* lst = list + (size_t)e * TOKENS;
    if constexpr (EPI == 0) {
        __bf16* outb = hr + (size_t)offsets[e] * (size_t)N;
        gemm_core<0, true>(Abase, B, M, K, N, lst, outb, nullptr, nullptr, nullptr);
    } else {
        const __bf16* A = Abase + (size_t)offsets[e] * (size_t)K;
        gemm_core<2, true>(A, B, M, K, N, nullptr, nullptr, outf, lst, wlist + (size_t)e * TOKENS);
    }
}

// ---------------------------------------------------------------------------
extern "C" void kernel_launch(void* const* d_in, const int* in_sizes, int n_in,
                              void* d_out, int out_size, void* d_ws, size_t ws_size,
                              hipStream_t stream) {
    const float* x      = (const float*)d_in[0];
    const float* gate_w = (const float*)d_in[1];
    const float* sw1    = (const float*)d_in[2];   // [2, D, H]
    const float* sw2    = (const float*)d_in[3];   // [2, H, D]
    const float* w1     = (const float*)d_in[4];   // [16, D, H]
    const float* w2     = (const float*)d_in[5];   // [16, H, D]
    float* out = (float*)d_out;
    float* aux = out + (size_t)(out_size - 1);

    char* p = (char*)d_ws;
    __bf16* xb = (__bf16*)p; p += (size_t)TOKENS * DIMSZ * 2;          // 16 MB
    __bf16* hs = (__bf16*)p; p += (size_t)2 * TOKENS * HIDSZ * 2;      // 16 MB
    __bf16* hr = (__bf16*)p; p += (size_t)2 * TOKENS * HIDSZ * 2;      // 16 MB
    float* probsum = (float*)p; p += 64;
    int*   count   = (int*)p;   p += 64;
    int*   offsets = (int*)p;   p += 64;
    int*   list    = (int*)p;   p += (size_t)NEXP * TOKENS * 4;
    float* wlist   = (float*)p; p += (size_t)NEXP * TOKENS * 4;

    dim3 blk(256);

    // 1) gating + routing lists + aux loss
    zero_kernel<<<1, 32, 0, stream>>>(probsum, count);
    gate_kernel<<<TOKENS / 8, blk, 0, stream>>>(x, gate_w, probsum, count, list, wlist);
    finalize_gate_kernel<<<1, 1, 0, stream>>>(probsum, count, offsets, aux);

    // 2) x -> bf16
    cvt_x_kernel<<<(TOKENS * DIMSZ) / (256 * 8), blk, 0, stream>>>(x, xb);

    // 3) shared experts
    dim3 g1(HIDSZ / BN, TOKENS / BM);
    for (int s = 0; s < 2; ++s)
        gemm_dense_kernel<0><<<g1, blk, 0, stream>>>(
            xb, sw1 + (size_t)s * DIMSZ * HIDSZ, TOKENS, DIMSZ, HIDSZ,
            hs + (size_t)s * TOKENS * HIDSZ, nullptr);
    dim3 g2(DIMSZ / BN, TOKENS / BM);
    gemm_dense_kernel<1><<<g2, blk, 0, stream>>>(
        hs, sw2, TOKENS, HIDSZ, DIMSZ, nullptr, out);
    gemm_dense_kernel<3><<<g2, blk, 0, stream>>>(
        hs + (size_t)TOKENS * HIDSZ, sw2 + (size_t)HIDSZ * DIMSZ,
        TOKENS, HIDSZ, DIMSZ, nullptr, out);

    // 4) routed experts: gather-GEMM-scatter
    dim3 gr1(HIDSZ / BN, TOKENS / BM, NEXP);
    gemm_routed_kernel<0><<<gr1, blk, 0, stream>>>(
        xb, w1, DIMSZ, HIDSZ, count, offsets, list, wlist, hr, nullptr);
    dim3 gr2(DIMSZ / BN, TOKENS / BM, NEXP);
    gemm_routed_kernel<2><<<gr2, blk, 0, stream>>>(
        hr, w2, HIDSZ, DIMSZ, count, offsets, list, wlist, hr, out);
}
